// GATSTEMEncoder_75720273428588
// MI455X (gfx1250) — compile-verified, matrix operand order checked by reading, not run on hardware
//
#include <hip/hip_runtime.h>
#include <hip/hip_bf16.h>

// ---------------------------------------------------------------------------
// GATSTEMEncoder for MI455X (gfx1250, wave32, WMMA).
//  - Dense GEMMs (x@W1, h@W2, h@Wf) use v_wmma_f32_16x16x32_bf16, f32 accum,
//    bf16 staged in LDS, register-staged software pipeline.
//    Requirements (satisfied by all call sites): K % 32 == 0, N % 128 == 0,
//    M >= 128.  M-tail handled by clamped loads + guarded stores.
//  - B tile staged column-per-thread so the transposed [n][k] LDS image is
//    written with two ds_store_b128 per thread (no b16 scatter).
//  - Attention path algebraically reduced: al_e = ea @ (We . a_edge) [16x4].
//  - Segment softmax: monotonic-uint atomicMax + atomicAdd exp-sum.
//  - Aggregation: per-edge scatter-add with f32 atomics.
// Workspace requirement: ~127 MB.
// ---------------------------------------------------------------------------

typedef __attribute__((ext_vector_type(16))) __bf16 v16bf;
typedef __attribute__((ext_vector_type(8)))  __bf16 v8bf;
typedef __attribute__((ext_vector_type(8)))  float  v8f;

#define BM 128
#define BN 128
#define BK 32
#define AS 40   // LDS row stride for A tile (bf16 units); 80 B, 16B-aligned rows
#define BS 40   // LDS row stride for B tile (bf16 units)

// ---------------------------------------------------------------------------
// Tiled GEMM: D[M,N] = A[M,K] @ B[K,N] (+bias).  A,B,D fp32 row-major.
// 256 threads = 8 waves (2x4); each wave computes 64x32 = 4x2 WMMA tiles.
// ---------------------------------------------------------------------------
__global__ __launch_bounds__(256)
void gemm_bf16_wmma(const float* __restrict__ A, const float* __restrict__ B,
                    const float* __restrict__ bias, float* __restrict__ D,
                    int M, int N, int K) {
  __shared__ __align__(16) __bf16 lsA[BM * AS];
  __shared__ __align__(16) __bf16 lsB[BN * BS];

  const int tid  = threadIdx.x;
  const int lane = tid & 31;
  const int wave = tid >> 5;
  const int wm   = wave >> 2;              // 0..1  (M direction)
  const int wn   = wave & 3;               // 0..3  (N direction)
  const int bM   = blockIdx.y * BM;
  const int bN   = blockIdx.x * BN;
  const int l15  = lane & 15;
  const int half = (lane & 16) ? 8 : 0;    // K-base select per ISA A/B layout

  // ---- A staging map: thread owns row rA, k-range kqA..kqA+15 -------------
  const int sA  = tid * 4;                 // first of 4 A float4 slots
  const int rA  = sA >> 3;                 // A row 0..127
  const int kqA = (sA & 7) << 2;           // A k-offset 0 or 16 (16 floats)
  const int grA = min(bM + rA, M - 1);     // clamped global row
  // ---- B staging map: thread owns column nB, k-range kqB..kqB+15 ----------
  const int nB  = tid & 127;               // local column 0..127
  const int kqB = (tid >> 7) << 4;         // 0 or 16
  const int gnB = min(bN + nB, N - 1);     // clamped global column

  v8f acc[4][2] = {};                      // f32 accumulators
  float4 aReg[4];
  float  bReg[16];

  // prologue fetch of tile 0 (K % BK == 0 assumed)
  {
    const float* ap = A + (size_t)grA * K + kqA;
    #pragma unroll
    for (int i = 0; i < 4; ++i) aReg[i] = *(const float4*)(ap + 4 * i);
    const float* bp = B + (size_t)kqB * N + gnB;
    #pragma unroll
    for (int j = 0; j < 16; ++j) bReg[j] = bp[(size_t)j * N];
  }

  for (int kb = 0; kb < K; kb += BK) {
    // ---- convert + store staged registers into LDS ----
    #pragma unroll
    for (int i = 0; i < 4; ++i) {
      union { __bf16 b[4]; uint2 u; } cv;
      cv.b[0] = (__bf16)aReg[i].x; cv.b[1] = (__bf16)aReg[i].y;
      cv.b[2] = (__bf16)aReg[i].z; cv.b[3] = (__bf16)aReg[i].w;
      *(uint2*)&lsA[rA * AS + kqA + 4 * i] = cv.u;
    }
    {
      union { __bf16 b[16]; uint4 u[2]; } cb;
      #pragma unroll
      for (int j = 0; j < 16; ++j) cb.b[j] = (__bf16)bReg[j];
      *(uint4*)&lsB[nB * BS + kqB]     = cb.u[0];
      *(uint4*)&lsB[nB * BS + kqB + 8] = cb.u[1];
    }
    __syncthreads();

    // ---- fetch next tile into registers (overlaps with WMMA below) ----
    const int kn = kb + BK;
    if (kn < K) {
      const float* ap = A + (size_t)grA * K + kn + kqA;
      #pragma unroll
      for (int i = 0; i < 4; ++i) aReg[i] = *(const float4*)(ap + 4 * i);
      const float* bp = B + (size_t)(kn + kqB) * N + gnB;
      #pragma unroll
      for (int j = 0; j < 16; ++j) bReg[j] = bp[(size_t)j * N];
      if (kn + BK < K) {   // keep L2 one more tile ahead
        __builtin_prefetch(A + (size_t)grA * K + kn + BK + kqA, 0, 1);
        __builtin_prefetch(B + (size_t)(kn + BK + kqB) * N + gnB, 0, 1);
      }
    }

    // ---- compute: fragments per ISA layout; element e -> K=(e<8?e:e+8)+half
    v16bf afrag[4];
    #pragma unroll
    for (int mt = 0; mt < 4; ++mt) {
      int row = wm * 64 + mt * 16 + l15;
      v8bf lo = *(const v8bf*)&lsA[row * AS + half];
      v8bf hi = *(const v8bf*)&lsA[row * AS + half + 16];
      afrag[mt] = __builtin_shufflevector(lo, hi, 0,1,2,3,4,5,6,7,8,9,10,11,12,13,14,15);
    }
    #pragma unroll
    for (int nt = 0; nt < 2; ++nt) {
      int col = wn * 32 + nt * 16 + l15;
      v8bf lo = *(const v8bf*)&lsB[col * BS + half];
      v8bf hi = *(const v8bf*)&lsB[col * BS + half + 16];
      v16bf bfrag = __builtin_shufflevector(lo, hi, 0,1,2,3,4,5,6,7,8,9,10,11,12,13,14,15);
      #pragma unroll
      for (int mt = 0; mt < 4; ++mt) {
        acc[mt][nt] = __builtin_amdgcn_wmma_f32_16x16x32_bf16(
            false, afrag[mt], false, bfrag, (short)0, acc[mt][nt], false, false);
      }
    }
    __syncthreads();
  }

  // ---- epilogue: C/D layout -> row = base + r + (lane>=16 ? 8:0), col = l15
  const int rh = (lane & 16) ? 8 : 0;
  #pragma unroll
  for (int mt = 0; mt < 4; ++mt) {
    #pragma unroll
    for (int nt = 0; nt < 2; ++nt) {
      int col = bN + wn * 32 + nt * 16 + l15;
      float bv = bias ? bias[col] : 0.f;
      #pragma unroll
      for (int r = 0; r < 8; ++r) {
        int row = bM + wm * 64 + mt * 16 + rh + r;
        if (row < M) {
          D[(size_t)row * N + col] = acc[mt][nt][r] + bv;
        }
      }
    }
  }
}

// ---------------------------------------------------------------------------
// Attention helpers
// ---------------------------------------------------------------------------
__device__ __forceinline__ unsigned encf(float f) {
  unsigned u = __float_as_uint(f);
  return (u & 0x80000000u) ? ~u : (u | 0x80000000u);
}
__device__ __forceinline__ float decf(unsigned u) {
  return (u & 0x80000000u) ? __uint_as_float(u & 0x7FFFFFFFu) : __uint_as_float(~u);
}

// ve[k*4+h] = sum_c We[k, h*256+c] * a_edge[h, c]   (16x4 precontraction)
__global__ void ve_kernel(const float* __restrict__ We, const float* __restrict__ ae,
                          float* __restrict__ ve) {
  int t = threadIdx.x;
  if (t >= 64) return;
  int k = t >> 2, h = t & 3;
  float s = 0.f;
  #pragma unroll 8
  for (int c = 0; c < 256; ++c) s += We[k * 1024 + h * 256 + c] * ae[h * 256 + c];
  ve[k * 4 + h] = s;
}

// degree + segment-sum of edge_attr per destination (fill_value='mean')
__global__ void deg_loop_accum(const int* __restrict__ dst, const float* __restrict__ ea,
                               float* __restrict__ loop_attr, float* __restrict__ deg, int E) {
  int i = blockIdx.x * blockDim.x + threadIdx.x;
  if (i >= E * 16) return;
  int e = i >> 4, d = i & 15;
  int n = dst[e];
  atomicAdd(&loop_attr[n * 16 + d], ea[(size_t)e * 16 + d]);
  if (d == 0) atomicAdd(&deg[n], 1.0f);
}

__global__ void loop_div(float* __restrict__ loop_attr, const float* __restrict__ deg, int Nn) {
  int i = blockIdx.x * blockDim.x + threadIdx.x;
  if (i >= Nn * 16) return;
  loop_attr[i] = loop_attr[i] / fmaxf(deg[i >> 4], 1.0f);
}

// al_src/al_dst: one wave per (node, head); C=256 dot products
__global__ __launch_bounds__(256)
void node_alpha(const float* __restrict__ xw, const float* __restrict__ asrc,
                const float* __restrict__ adst, float* __restrict__ al_src,
                float* __restrict__ al_dst, int Nn) {
  int wave = threadIdx.x >> 5, lane = threadIdx.x & 31;
  int idx = blockIdx.x * 8 + wave;
  if (idx >= Nn * 4) return;
  int n = idx >> 2, h = idx & 3;
  const float* xp = xw + (size_t)n * 1024 + h * 256;
  const float* ap = asrc + h * 256;
  const float* bp = adst + h * 256;
  float s1 = 0.f, s2 = 0.f;
  for (int c = lane; c < 256; c += 32) {
    float x = xp[c];
    s1 += x * ap[c];
    s2 += x * bp[c];
  }
  #pragma unroll
  for (int off = 16; off > 0; off >>= 1) {
    s1 += __shfl_down(s1, off, 32);
    s2 += __shfl_down(s2, off, 32);
  }
  if (lane == 0) { al_src[idx] = s1; al_dst[idx] = s2; }
}

// alpha = leaky_relu(al_src[src]+al_dst[dst]+al_e); atomic segment max
__global__ void edge_alpha_max(const int* __restrict__ src, const int* __restrict__ dst,
                               const float* __restrict__ ea, const float* __restrict__ loop_attr,
                               const float* __restrict__ ve, const float* __restrict__ al_src,
                               const float* __restrict__ al_dst, float* __restrict__ alphabuf,
                               unsigned* __restrict__ menc, int E, int Nn) {
  int i = blockIdx.x * blockDim.x + threadIdx.x;
  int E2 = E + Nn;
  if (i >= E2 * 4) return;
  int e = i >> 2, h = i & 3;
  int sv, dv;
  const float* eap;
  if (e < E) { sv = src[e]; dv = dst[e]; eap = ea + (size_t)e * 16; }
  else       { sv = dv = e - E;          eap = loop_attr + (size_t)(e - E) * 16; }
  float ale = 0.f;
  #pragma unroll
  for (int k = 0; k < 16; ++k) ale += eap[k] * ve[k * 4 + h];
  float a = al_src[sv * 4 + h] + al_dst[dv * 4 + h] + ale;
  a = a > 0.f ? a : 0.2f * a;
  alphabuf[i] = a;
  atomicMax(&menc[dv * 4 + h], encf(a));
}

// e = exp(alpha - m[dst]); atomic segment sum
__global__ void edge_expsum(const int* __restrict__ dst, float* __restrict__ alphabuf,
                            const unsigned* __restrict__ menc, float* __restrict__ ssum,
                            int E, int Nn) {
  int i = blockIdx.x * blockDim.x + threadIdx.x;
  int E2 = E + Nn;
  if (i >= E2 * 4) return;
  int e = i >> 2, h = i & 3;
  int dv = (e < E) ? dst[e] : (e - E);
  float ex = __expf(alphabuf[i] - decf(menc[dv * 4 + h]));
  alphabuf[i] = ex;
  atomicAdd(&ssum[dv * 4 + h], ex);
}

// out[dst] += (e/s[dst]) * xw[src]; one block per edge, 1024 channels
__global__ __launch_bounds__(256)
void aggregate(const int* __restrict__ src, const int* __restrict__ dst,
               const float* __restrict__ alphabuf, const float* __restrict__ ssum,
               const float* __restrict__ xw, float* __restrict__ agg, int E, int Nn) {
  int e = blockIdx.x;
  int sv, dv;
  if (e < E) { sv = src[e]; dv = dst[e]; } else { sv = dv = e - E; }
  int j = threadIdx.x * 4;
  int h = j >> 8;
  float w = alphabuf[(size_t)e * 4 + h] / (ssum[dv * 4 + h] + 1e-16f);
  const float* xp = xw + (size_t)sv * 1024 + j;
  float* op = agg + (size_t)dv * 1024 + j;
  float4 x = *(const float4*)xp;
  atomicAdd(op + 0, w * x.x);
  atomicAdd(op + 1, w * x.y);
  atomicAdd(op + 2, w * x.z);
  atomicAdd(op + 3, w * x.w);
}

// h = elu(agg + b)
__global__ void bias_elu(const float* __restrict__ agg, const float* __restrict__ b,
                         float* __restrict__ h, int total, int ncols) {
  int i = blockIdx.x * blockDim.x + threadIdx.x;
  if (i >= total) return;
  float v = agg[i] + b[i % ncols];
  h[i] = v > 0.f ? v : (__expf(v) - 1.f);
}

// ---------------------------------------------------------------------------
// Host orchestration
// ---------------------------------------------------------------------------
static inline char* ws_take(char*& p, size_t bytes) {
  char* r = p;
  p += (bytes + 255) & ~(size_t)255;
  return r;
}

extern "C" void kernel_launch(void* const* d_in, const int* in_sizes, int n_in,
                              void* d_out, int out_size, void* d_ws, size_t ws_size,
                              hipStream_t stream) {
  const float* x   = (const float*)d_in[0];
  const int*   ei  = (const int*)d_in[1];
  const float* ef  = (const float*)d_in[2];
  const float* W1  = (const float*)d_in[3];
  const float* as1 = (const float*)d_in[4];
  const float* ad1 = (const float*)d_in[5];
  const float* ae1 = (const float*)d_in[6];
  const float* We1 = (const float*)d_in[7];
  const float* b1  = (const float*)d_in[8];
  const float* W2  = (const float*)d_in[9];
  const float* as2 = (const float*)d_in[10];
  const float* ad2 = (const float*)d_in[11];
  const float* ae2 = (const float*)d_in[12];
  const float* We2 = (const float*)d_in[13];
  const float* b2  = (const float*)d_in[14];
  const float* Wf  = (const float*)d_in[15];
  const float* bfv = (const float*)d_in[16];

  const int Nn = in_sizes[0] / 128;   // 10000
  const int E  = in_sizes[1] / 2;     // 160000
  const int E2 = E + Nn;
  const int HC = 1024;
  const int NF = in_sizes[16];        // 25088

  // workspace layout (~127 MB)
  char* wp = (char*)d_ws;
  float*    XW    = (float*)   ws_take(wp, (size_t)Nn * HC * 4);
  float*    Hb    = (float*)   ws_take(wp, (size_t)Nn * HC * 4);
  float*    AGG   = (float*)   ws_take(wp, (size_t)Nn * HC * 4);
  float*    AL    = (float*)   ws_take(wp, (size_t)E2 * 4 * 4);   // alphabuf
  float*    ALS   = (float*)   ws_take(wp, (size_t)Nn * 4 * 4);
  float*    ALD   = (float*)   ws_take(wp, (size_t)Nn * 4 * 4);
  unsigned* MEN   = (unsigned*)ws_take(wp, (size_t)Nn * 4 * 4);
  float*    SSUM  = (float*)   ws_take(wp, (size_t)Nn * 4 * 4);
  float*    DEG   = (float*)   ws_take(wp, (size_t)Nn * 4);
  float*    LOOPA = (float*)   ws_take(wp, (size_t)Nn * 16 * 4);
  float*    VE    = (float*)   ws_take(wp, 16 * 4 * 4);

  const int* srcp = ei;
  const int* dstp = ei + E;

  const int T = 256;
  dim3 blk(T);

  // ---- self-loop attr (shared by both layers) ----
  hipMemsetAsync(DEG,   0, (size_t)Nn * 4, stream);
  hipMemsetAsync(LOOPA, 0, (size_t)Nn * 16 * 4, stream);
  deg_loop_accum<<<(E * 16 + T - 1) / T, blk, 0, stream>>>(dstp, ef, LOOPA, DEG, E);
  loop_div<<<(Nn * 16 + T - 1) / T, blk, 0, stream>>>(LOOPA, DEG, Nn);

  dim3 g_gemm1(HC / BN, (Nn + BM - 1) / BM);            // 8 x 79
  dim3 g_gemmF((NF + BN - 1) / BN, (Nn + BM - 1) / BM); // 196 x 79
  int  g_edge = (E2 * 4 + T - 1) / T;

  // ================= layer 1 =================
  gemm_bf16_wmma<<<g_gemm1, blk, 0, stream>>>(x, W1, nullptr, XW, Nn, HC, 128);
  ve_kernel<<<1, 64, 0, stream>>>(We1, ae1, VE);
  node_alpha<<<(Nn * 4 + 7) / 8, blk, 0, stream>>>(XW, as1, ad1, ALS, ALD, Nn);
  hipMemsetAsync(MEN,  0, (size_t)Nn * 4 * 4, stream);
  hipMemsetAsync(SSUM, 0, (size_t)Nn * 4 * 4, stream);
  edge_alpha_max<<<g_edge, blk, 0, stream>>>(srcp, dstp, ef, LOOPA, VE, ALS, ALD, AL, MEN, E, Nn);
  edge_expsum<<<g_edge, blk, 0, stream>>>(dstp, AL, MEN, SSUM, E, Nn);
  hipMemsetAsync(AGG, 0, (size_t)Nn * HC * 4, stream);
  aggregate<<<E2, blk, 0, stream>>>(srcp, dstp, AL, SSUM, XW, AGG, E, Nn);
  bias_elu<<<(Nn * HC + T - 1) / T, blk, 0, stream>>>(AGG, b1, Hb, Nn * HC, HC);

  // ================= layer 2 =================
  gemm_bf16_wmma<<<g_gemm1, blk, 0, stream>>>(Hb, W2, nullptr, XW, Nn, HC, HC);
  ve_kernel<<<1, 64, 0, stream>>>(We2, ae2, VE);
  node_alpha<<<(Nn * 4 + 7) / 8, blk, 0, stream>>>(XW, as2, ad2, ALS, ALD, Nn);
  hipMemsetAsync(MEN,  0, (size_t)Nn * 4 * 4, stream);
  hipMemsetAsync(SSUM, 0, (size_t)Nn * 4 * 4, stream);
  edge_alpha_max<<<g_edge, blk, 0, stream>>>(srcp, dstp, ef, LOOPA, VE, ALS, ALD, AL, MEN, E, Nn);
  edge_expsum<<<g_edge, blk, 0, stream>>>(dstp, AL, MEN, SSUM, E, Nn);
  hipMemsetAsync(AGG, 0, (size_t)Nn * HC * 4, stream);
  aggregate<<<E2, blk, 0, stream>>>(srcp, dstp, AL, SSUM, XW, AGG, E, Nn);
  bias_elu<<<(Nn * HC + T - 1) / T, blk, 0, stream>>>(AGG, b2, Hb, Nn * HC, HC);

  // ================= final linear =================
  gemm_bf16_wmma<<<g_gemmF, blk, 0, stream>>>(Hb, Wf, bfv, (float*)d_out, Nn, NF, HC);
}
